// QuantizedLinear_87007447482449
// MI455X (gfx1250) — matrix operand order repaired and checked by reading, not compiled, not run on hardware
//
#include <hip/hip_runtime.h>

typedef __attribute__((ext_vector_type(16))) __bf16 bf16x16;
typedef __attribute__((ext_vector_type(8)))  __bf16 bf16x8;
typedef __attribute__((ext_vector_type(8)))  float  f32x8;

constexpr int BROWS = 8192;
constexpr int INF   = 4096;
constexpr int OUTF  = 4096;

constexpr int BM = 256, BN = 128, BK = 32;
constexpr int LDA = BK + 8;    // 40 bf16 = 80B row stride (16B-aligned b128, bank-spread)
constexpr int LDB = BK + 8;    // 40 bf16
constexpr int LDW = BK + 16;   // 48B raw int8 row stride (16B-aligned)
constexpr int NK  = INF / BK;  // 128 K-steps
constexpr int BDIM = 256;      // 8 waves: 4 (M) x 2 (N), each owns 64x64 -> 16 WMMA/step

__global__ __launch_bounds__(BDIM) void qlinear_bf16_wmma(
    const float* __restrict__ X,
    const signed char* __restrict__ W,
    const float* __restrict__ bias,
    const float* __restrict__ scale_p,
    const signed char* __restrict__ zp_p,
    float* __restrict__ out)
{
    __shared__ __align__(16) __bf16      As[2][BM * LDA];   // 40 KB
    __shared__ __align__(16) __bf16      Bs[2][BN * LDB];   // 20 KB
    __shared__ __align__(16) signed char Wraw[2][BN * LDW]; // 12 KB

    const int tid   = threadIdx.x;
    const int lane  = tid & 31;
    const int wid   = tid >> 5;
    const int waveM = wid >> 1;   // 0..3 -> 64-row slab
    const int waveN = wid & 1;    // 0..1 -> 64-col slab
    const int l16   = lane & 15;
    const int g     = lane >> 4;

    const int gm0 = blockIdx.y * BM;
    const int gn0 = blockIdx.x * BN;

    const float scale = scale_p[0];
    const int   zp    = (int)zp_p[0];
    // magic-number dequant: bitcast(0x4B000000|(b^0x80)) = 2^23 + (w+128)
    const float dqbias = 8388608.0f + 128.0f + (float)zp;

    // A loader: one full 32-float row slice per thread
    const int ar = tid;                        // 0..255
    const float* xrow = X + (size_t)(gm0 + ar) * INF;

    // W loader: 16 bytes per thread, 2 threads per row
    const int wr = tid >> 1;                   // 0..127
    const int wc = (tid & 1) * 16;             // 0 or 16
    const signed char* wbase = W + (size_t)gn0 * INF;
    const unsigned wlaneoff  = (unsigned)(wr * INF + wc);
    const unsigned ldsW0 = (unsigned)(uintptr_t)&Wraw[0][wr * LDW + wc];
    const unsigned ldsW1 = (unsigned)(uintptr_t)&Wraw[1][wr * LDW + wc];

    f32x8  acc[4][4] = {};
    float4 aReg[8];

    // ---- prologue: K-tile 0 -> A regs + async W copy into Wraw[0]
    {
        const float4* xp = (const float4*)xrow;
#pragma unroll
        for (int u = 0; u < 8; ++u) aReg[u] = xp[u];
        asm volatile("global_load_async_to_lds_b128 %0, %1, %2"
                     :: "v"(ldsW0), "v"(wlaneoff), "s"(wbase) : "memory");
    }

    for (int kt = 0; kt < NK; ++kt) {
        const int buf = kt & 1;

        // async W tile for this step has landed?
        asm volatile("s_wait_asynccnt 0x0" ::: "memory");
        __syncthreads();

        // ---- stage A: f32 regs -> bf16 LDS (v_cvt_pk_bf16_f32 pairs)
        {
            __bf16* dA = &As[buf][ar * LDA];
            const float* af = (const float*)aReg;
#pragma unroll
            for (int q = 0; q < 4; ++q) {
                bf16x8 p;
#pragma unroll
                for (int e = 0; e < 8; ++e) p[e] = (__bf16)af[q * 8 + e];
                *(bf16x8*)(dA + q * 8) = p;
            }
        }
        // ---- stage W: raw int8 LDS -> (w - zp) exact bf16 LDS, ~2.5 VALU/byte
        {
            const uint4 w4 = *(const uint4*)&Wraw[buf][wr * LDW + wc];
            unsigned uw[4] = { w4.x ^ 0x80808080u, w4.y ^ 0x80808080u,
                               w4.z ^ 0x80808080u, w4.w ^ 0x80808080u };
            __bf16* dB = &Bs[buf][wr * LDB + wc];
#pragma unroll
            for (int q = 0; q < 2; ++q) {
                bf16x8 p;
#pragma unroll
                for (int e = 0; e < 8; ++e) {
                    const unsigned word = uw[q * 2 + (e >> 2)];
                    const unsigned fb = __builtin_amdgcn_perm(
                        0x4B000000u, word, 0x07050400u + (unsigned)(e & 3));
                    p[e] = (__bf16)(__uint_as_float(fb) - dqbias);
                }
                *(bf16x8*)(dB + q * 8) = p;
            }
        }
        __syncthreads();

        // ---- fetch K-tile kt+1; latency overlaps the 16-WMMA burst
        if (kt + 1 < NK) {
            const float4* xp = (const float4*)(xrow + (kt + 1) * BK);
#pragma unroll
            for (int u = 0; u < 8; ++u) aReg[u] = xp[u];
            const unsigned dst = (buf ? ldsW0 : ldsW1);
            const unsigned off = wlaneoff + (unsigned)((kt + 1) * BK);
            asm volatile("global_load_async_to_lds_b128 %0, %1, %2"
                         :: "v"(dst), "v"(off), "s"(wbase) : "memory");
        }
        if (kt + 2 < NK)
            __builtin_prefetch(xrow + (kt + 2) * BK, 0, 0);

        // ---- fragments per ISA VGPR layouts
        bf16x16 afrag[4], bfrag[4];
#pragma unroll
        for (int i = 0; i < 4; ++i) {
            const __bf16* p = &As[buf][(waveM * 64 + i * 16 + l16) * LDA + g * 8];
            bf16x8 lo = *(const bf16x8*)p;
            bf16x8 hi = *(const bf16x8*)(p + 16);
            afrag[i] = __builtin_shufflevector(lo, hi,
                0,1,2,3,4,5,6,7,8,9,10,11,12,13,14,15);
        }
#pragma unroll
        for (int j = 0; j < 4; ++j) {
            const __bf16* q = &Bs[buf][(waveN * 64 + j * 16 + l16) * LDB + g * 8];
            bf16x8 lo = *(const bf16x8*)q;
            bf16x8 hi = *(const bf16x8*)(q + 16);
            bfrag[j] = __builtin_shufflevector(lo, hi,
                0,1,2,3,4,5,6,7,8,9,10,11,12,13,14,15);
        }

        // ---- 16x v_wmma_f32_16x16x32_bf16 per K-step per wave
#pragma unroll
        for (int i = 0; i < 4; ++i)
#pragma unroll
            for (int j = 0; j < 4; ++j)
                acc[i][j] = __builtin_amdgcn_wmma_f32_16x16x32_bf16(
                    false, afrag[i], false, bfrag[j],
                    (short)0, acc[i][j], false, false);
    }

    // ---- epilogue: out = scale*acc + bias   (C/D layout: N=l16, M=g*8+r)
#pragma unroll
    for (int j = 0; j < 4; ++j) {
        const int n = gn0 + waveN * 64 + j * 16 + l16;
        const float bj = bias[n];
#pragma unroll
        for (int i = 0; i < 4; ++i) {
            const int m0 = gm0 + waveM * 64 + i * 16 + g * 8;
            float* o = out + (size_t)m0 * OUTF + n;
#pragma unroll
            for (int r = 0; r < 8; ++r)
                o[(size_t)r * OUTF] = acc[i][j][r] * scale + bj;
        }
    }
}

extern "C" void kernel_launch(void* const* d_in, const int* in_sizes, int n_in,
                              void* d_out, int out_size, void* d_ws, size_t ws_size,
                              hipStream_t stream) {
    const float*       X     = (const float*)d_in[0];
    const signed char* W     = (const signed char*)d_in[1];
    const float*       bias  = (const float*)d_in[2];
    const float*       scale = (const float*)d_in[3];
    const signed char* zp    = (const signed char*)d_in[4];
    float*             out   = (float*)d_out;

    dim3 grid(OUTF / BN, BROWS / BM);   // (32, 32)
    qlinear_bf16_wmma<<<grid, BDIM, 0, stream>>>(X, W, bias, scale, zp, out);
}
